// AttentionQKV_21122649162321
// MI455X (gfx1250) — matrix operand order repaired and checked by reading, not compiled, not run on hardware
//
#include <hip/hip_runtime.h>
#include <hip/hip_bf16.h>

typedef _Float16 half_t;
typedef __attribute__((ext_vector_type(16))) _Float16 v16h;
typedef __attribute__((ext_vector_type(8)))  _Float16 v8h;
typedef __attribute__((ext_vector_type(8)))  float    v8f;

#define B_   2
#define N_   2048
#define DIM_ 1024
#define H_   16
#define D_   64
#define BH_  (B_*H_)
#define BN_  (B_*N_)

// ------------------------- generic f16 WMMA GEMM -------------------------
// Block tile 256x64, 256 threads = 8 wave32 waves, each wave -> 32 rows x 64 cols
// (8 v_wmma_f32_16x16x32_f16 per K-step). Double-buffered LDS fed by
// global_load_async_to_lds_b128 (ASYNCcnt), one barrier per K-step.
// C = alpha * A*B + bias + addend.
//   A: [M x K] f16 row-major (lda); B: [N x K] f16 row-major (ldb)  (K-contiguous!)
// Batch offset for A/B: (z/c)*outer + (z%c)*inner.
#define TM 256
#define TN 64
#define TK 32
#define LDS_S 40   // padded LDS row stride in halves

__global__ __launch_bounds__(256) void gemm_f16_wmma(
    const half_t* __restrict__ A, long long sAo, long long sAi, int cA, int lda,
    const half_t* __restrict__ Bm, long long sBo, long long sBi, int cB, int ldb,
    float* Cf, half_t* Ch, long long sC, int ldc, int storeT,
    const float* __restrict__ bias,
    const float* addend, long long sAdd, int ldadd,
    int K, float alpha, int causalSkip)
{
    const int z  = blockIdx.z;
    const int m0 = blockIdx.y * TM;
    const int n0 = blockIdx.x * TN;
    if (causalSkip && n0 > m0 + (TM - 1)) return;   // block fully above causal diagonal

    const half_t* Ab = A  + (long long)(z / cA) * sAo + (long long)(z % cA) * sAi;
    const half_t* Bb = Bm + (long long)(z / cB) * sBo + (long long)(z % cB) * sBi;

    __shared__ half_t As[2][TM * LDS_S];
    __shared__ half_t Bs[2][TN * LDS_S];

    const int t     = threadIdx.x;
    const int w     = t >> 5;         // wave id 0..7
    const int lane  = t & 31;
    const int lhalf = lane >> 4;      // 0 or 1
    const int l15   = lane & 15;

    // issue async copies for one K-slab into LDS buffer `buf`
    auto stage = [&](int buf, int k0) {
        unsigned aBase = (unsigned)(size_t)(&As[buf][0]);
        unsigned bBase = (unsigned)(size_t)(&Bs[buf][0]);
        #pragma unroll
        for (int i = 0; i < 4; ++i) {               // A: 256x32 halves = 1024 16B chunks
            int cidx = t + 256 * i;
            int r    = cidx >> 2;
            int kc   = (cidx & 3) * 8;
            const half_t* g = Ab + (long long)(m0 + r) * lda + (k0 + kc);
            unsigned loff = aBase + (unsigned)(r * LDS_S + kc) * 2u;
            asm volatile("global_load_async_to_lds_b128 %0, %1, off"
                         :: "v"(loff), "v"(g) : "memory");
        }
        {                                           // B: 64x32 halves = 256 16B chunks
            int n  = t >> 2;
            int kc = (t & 3) * 8;
            const half_t* g = Bb + (long long)(n0 + n) * ldb + (k0 + kc);
            unsigned loff = bBase + (unsigned)(n * LDS_S + kc) * 2u;
            asm volatile("global_load_async_to_lds_b128 %0, %1, off"
                         :: "v"(loff), "v"(g) : "memory");
        }
    };

    v8f c[2][4] = {};
    stage(0, 0);
    int buf = 0;
    for (int k0 = 0; k0 < K; k0 += TK) {
        asm volatile("s_wait_asynccnt 0" ::: "memory");
        __syncthreads();
        if (k0 + TK < K) stage(buf ^ 1, k0 + TK);

        const int koA = lhalf * 8;       // A frag: lanes>=16 hold K+8
        const int koB = lhalf * 8;       // B frag: lanes>=16 hold K+8
        union Frag { v16h v; v8h h[2]; };
        Frag afr[2];
        #pragma unroll
        for (int mt = 0; mt < 2; ++mt) {
            const int arow = 32 * w + 16 * mt + l15;
            afr[mt].h[0] = *reinterpret_cast<const v8h*>(&As[buf][arow * LDS_S + koA]);
            afr[mt].h[1] = *reinterpret_cast<const v8h*>(&As[buf][arow * LDS_S + koA + 16]);
        }
        #pragma unroll
        for (int nt = 0; nt < 4; ++nt) {
            Frag bfr;
            const int bcol = nt * 16 + l15;
            bfr.h[0] = *reinterpret_cast<const v8h*>(&Bs[buf][bcol * LDS_S + koB]);
            bfr.h[1] = *reinterpret_cast<const v8h*>(&Bs[buf][bcol * LDS_S + koB + 16]);
            #pragma unroll
            for (int mt = 0; mt < 2; ++mt)
                c[mt][nt] = __builtin_amdgcn_wmma_f32_16x16x32_f16(
                                false, afr[mt].v, false, bfr.v, (short)0,
                                c[mt][nt], false, false);
        }
        buf ^= 1;
    }

    // epilogue: C layout -> lane l, vgpr r holds (M = r + 8*(l>=16), N = l&15)
    const long long co = (long long)z * sC;
    const int colb = n0 + l15;
    #pragma unroll
    for (int mt = 0; mt < 2; ++mt) {
        const int rowb = m0 + 32 * w + 16 * mt + 8 * lhalf;
        #pragma unroll
        for (int nt = 0; nt < 4; ++nt) {
            const int col = colb + nt * 16;
            const float bv = bias ? bias[col] : 0.0f;
            #pragma unroll
            for (int r = 0; r < 8; ++r) {
                const int row = rowb + r;
                float val = alpha * c[mt][nt][r] + bv;
                if (addend) val += addend[(long long)z * sAdd + (long long)row * ldadd + col];
                const long long idx = storeT ? (co + (long long)col * ldc + row)
                                             : (co + (long long)row * ldc + col);
                if (Ch) Ch[idx] = (half_t)val;
                else    Cf[idx] = val;
            }
        }
    }
}

// ------------------------- helpers -------------------------
// dst [N x K] f16 = transpose(src [K x N] f32)
__global__ __launch_bounds__(256) void cast_transpose_f32_f16(
    const float* __restrict__ s, half_t* __restrict__ d, int Kdim, int Ndim) {
    long long i = (long long)blockIdx.x * 256 + threadIdx.x;
    if (i >= (long long)Kdim * Ndim) return;
    int k = (int)(i % Kdim);
    int n = (int)(i / Kdim);
    d[i] = (half_t)s[(long long)k * Ndim + n];
}

template <typename OutT>
__global__ __launch_bounds__(256) void layernorm_row(const float* __restrict__ x,
                                                     const float* __restrict__ gamma,
                                                     OutT* __restrict__ out, int dim) {
    const long long row = blockIdx.x;
    const float* xr = x + row * dim;
    OutT* orow = out + row * dim;
    __shared__ float red[256];
    const int t = threadIdx.x;
    float s = 0.f;
    for (int j = t; j < dim; j += 256) s += xr[j];
    red[t] = s; __syncthreads();
    for (int o = 128; o; o >>= 1) { if (t < o) red[t] += red[t + o]; __syncthreads(); }
    const float mu = red[0] / dim;
    __syncthreads();
    float v = 0.f;
    for (int j = t; j < dim; j += 256) { float d = xr[j] - mu; v += d * d; }
    red[t] = v; __syncthreads();
    for (int o = 128; o; o >>= 1) { if (t < o) red[t] += red[t + o]; __syncthreads(); }
    const float inv = rsqrtf(red[0] / dim + 1e-5f);
    for (int j = t; j < dim; j += 256) orow[j] = (OutT)((xr[j] - mu) * inv * gamma[j]);
}

// causal row softmax: P[i,j] = softmax_{j<=i}(S[i,:]), zeros above diag.
// deg[i] = 2 - P[i,i]  (rowsum of adj = clip(attn + I)).
__global__ __launch_bounds__(256) void softmax_causal(const half_t* __restrict__ S,
                                                      half_t* __restrict__ P,
                                                      float* __restrict__ deg, int n) {
    const long long row = blockIdx.x;          // bh*N + i
    const int i = (int)(row % n);
    const half_t* s = S + row * n;
    half_t* p = P + row * n;
    __shared__ float red[256];
    const int t = threadIdx.x;
    float mx = -3.0e38f;
    for (int j = t; j <= i; j += 256) mx = fmaxf(mx, (float)s[j]);
    red[t] = mx; __syncthreads();
    for (int o = 128; o; o >>= 1) { if (t < o) red[t] = fmaxf(red[t], red[t + o]); __syncthreads(); }
    mx = red[0]; __syncthreads();
    float sum = 0.f;
    for (int j = t; j <= i; j += 256) sum += __expf((float)s[j] - mx);
    red[t] = sum; __syncthreads();
    for (int o = 128; o; o >>= 1) { if (t < o) red[t] += red[t + o]; __syncthreads(); }
    const float inv = 1.0f / red[0];
    for (int j = t; j < n; j += 256) {
        float pv = (j <= i) ? __expf((float)s[j] - mx) * inv : 0.0f;
        p[j] = (half_t)pv;
        if (j == i) deg[row] = 2.0f - pv;
    }
}

// feats[bh,n,d]  <-  src[(b,n),(h,d)]
__global__ __launch_bounds__(256) void split_heads(const half_t* __restrict__ src,
                                                   half_t* __restrict__ dst) {
    long long i = (long long)blockIdx.x * 256 + threadIdx.x;   // over BH*N*D
    if (i >= (long long)BH_ * N_ * D_) return;
    int d = (int)(i % D_);  long long t1 = i / D_;
    int n = (int)(t1 % N_); long long t2 = t1 / N_;
    int h = (int)(t2 % H_); int b = (int)(t2 / H_);
    dst[i] = src[(((long long)b * N_ + n) * H_ + h) * D_ + d];
}

// vT[bh,d,n]  <-  src[(b,n),(h,d)]   (transposed per-head view for B operands)
__global__ __launch_bounds__(256) void split_heads_T(const half_t* __restrict__ src,
                                                     half_t* __restrict__ dst) {
    long long i = (long long)blockIdx.x * 256 + threadIdx.x;   // over BH*D*N
    if (i >= (long long)BH_ * D_ * N_) return;
    int n = (int)(i % N_);  long long t1 = i / N_;
    int d = (int)(t1 % D_); long long t2 = t1 / D_;
    int h = (int)(t2 % H_); int b = (int)(t2 / H_);
    dst[i] = src[(((long long)b * N_ + n) * H_ + h) * D_ + d];
}

// merged[(b,n),(h,d)] (f16) <- src[bh,n,d] (f32)
__global__ __launch_bounds__(256) void merge_heads(const float* __restrict__ src,
                                                   half_t* __restrict__ dst) {
    long long i = (long long)blockIdx.x * 256 + threadIdx.x;   // over BN*H*D
    if (i >= (long long)BN_ * H_ * D_) return;
    int d = (int)(i % D_);  long long t1 = i / D_;
    int h = (int)(t1 % H_); long long t2 = t1 / H_;
    int n = (int)(t2 % N_); int b = (int)(t2 / N_);
    dst[i] = (half_t)src[(((long long)b * H_ + h) * N_ + n) * D_ + d];
}

// feats = gelu( (G + (deg-1)*h)/deg + feats );  hT is [bh, D, N] (transposed)
__global__ __launch_bounds__(256) void gnn_pointwise(const float* __restrict__ G,
                                                     const half_t* __restrict__ hT,
                                                     const float* __restrict__ deg,
                                                     half_t* feats) {
    long long i = (long long)blockIdx.x * 256 + threadIdx.x;   // over BH*N*D
    if (i >= (long long)BH_ * N_ * D_) return;
    const int d = (int)(i % D_);
    const long long nrow = i / D_;               // bh*N + n
    const int n = (int)(nrow % N_);
    const long long bh = nrow / N_;
    const float dg = deg[nrow];
    const float hv = (float)hT[(bh * D_ + d) * N_ + n];
    const float hn = (G[i] + (dg - 1.0f) * hv) / dg;
    const float xs = hn + (float)feats[i];
    feats[i] = (half_t)(0.5f * xs * (1.0f + erff(xs * 0.70710678118f)));
}

// ------------------------- launch -------------------------
extern "C" void kernel_launch(void* const* d_in, const int* in_sizes, int n_in,
                              void* d_out, int out_size, void* d_ws, size_t ws_size,
                              hipStream_t stream) {
    const float* x        = (const float*)d_in[0];
    const float* gamma_in = (const float*)d_in[1];
    const float* Wq       = (const float*)d_in[2];
    const float* Wk       = (const float*)d_in[3];
    const float* Wv       = (const float*)d_in[4];
    const float* gnn_w    = (const float*)d_in[5];
    const float* gnn_b    = (const float*)d_in[6];
    const float* lin_w    = (const float*)d_in[7];
    const float* lin_b    = (const float*)d_in[8];
    const float* Wo       = (const float*)d_in[9];
    const float* gamma_out= (const float*)d_in[10];
    float* out = (float*)d_out;

    // workspace layout (256B aligned)
    char* w = (char*)d_ws;
    long long off = 0;
    auto nxt = [&](long long sz) { long long r = off; off += (sz + 255) & ~255LL; return r; };
    half_t* wqT   = (half_t*)(w + nxt((long long)DIM_ * DIM_ * 2));   // [N x K] transposed
    half_t* wkT   = (half_t*)(w + nxt((long long)DIM_ * DIM_ * 2));
    half_t* wvT   = (half_t*)(w + nxt((long long)DIM_ * DIM_ * 2));
    half_t* woT   = (half_t*)(w + nxt((long long)DIM_ * DIM_ * 2));
    half_t* gwT   = (half_t*)(w + nxt(2LL * D_ * D_ * 2));
    half_t* lwT   = (half_t*)(w + nxt((long long)D_ * D_ * 2));
    half_t* xn    = (half_t*)(w + nxt((long long)BN_ * DIM_ * 2));
    half_t* qt    = (half_t*)(w + nxt((long long)BN_ * DIM_ * 2));
    half_t* kt    = (half_t*)(w + nxt((long long)BN_ * DIM_ * 2));
    half_t* vt    = (half_t*)(w + nxt((long long)BN_ * DIM_ * 2));
    half_t* vT    = (half_t*)(w + nxt((long long)BH_ * D_ * N_ * 2));
    half_t* Sh    = (half_t*)(w + nxt((long long)BH_ * N_ * N_ * 2));
    half_t* Ph    = (half_t*)(w + nxt((long long)BH_ * N_ * N_ * 2));
    float*  deg   = (float*) (w + nxt((long long)BH_ * N_ * 4));
    float*  ao    = (float*) (w + nxt((long long)BH_ * N_ * D_ * 4));
    half_t* feats = (half_t*)(w + nxt((long long)BH_ * N_ * D_ * 2));
    half_t* hT    = (half_t*)(w + nxt((long long)BH_ * D_ * N_ * 2));
    float*  Gbuf  = (float*) (w + nxt((long long)BH_ * N_ * D_ * 4));
    half_t* merged= (half_t*)(w + nxt((long long)BN_ * DIM_ * 2));
    float*  Y     = (float*) (w + nxt((long long)BN_ * DIM_ * 4));
    if ((size_t)off > ws_size) return;   // insufficient scratch: deterministic no-op

    const long long sQKV = (long long)N_ * DIM_;  // per-b stride in q/k/v temp
    const long long sNN  = (long long)N_ * N_;
    const long long sND  = (long long)N_ * D_;
    const dim3 blk(256);
    const int nElemPH = (int)((long long)BH_ * N_ * D_ / 256);

    // weights -> transposed f16 (B operands are [N x K], K-contiguous)
    cast_transpose_f32_f16<<<(DIM_*DIM_ + 255)/256, blk, 0, stream>>>(Wq, wqT, DIM_, DIM_);
    cast_transpose_f32_f16<<<(DIM_*DIM_ + 255)/256, blk, 0, stream>>>(Wk, wkT, DIM_, DIM_);
    cast_transpose_f32_f16<<<(DIM_*DIM_ + 255)/256, blk, 0, stream>>>(Wv, wvT, DIM_, DIM_);
    cast_transpose_f32_f16<<<(DIM_*DIM_ + 255)/256, blk, 0, stream>>>(Wo, woT, DIM_, DIM_);
    cast_transpose_f32_f16<<<(D_*D_ + 255)/256, blk, 0, stream>>>(gnn_w,           gwT,           D_, D_);
    cast_transpose_f32_f16<<<(D_*D_ + 255)/256, blk, 0, stream>>>(gnn_w + D_*D_,   gwT + D_*D_,   D_, D_);
    cast_transpose_f32_f16<<<(D_*D_ + 255)/256, blk, 0, stream>>>(lin_w, lwT, D_, D_);

    // input layernorm -> f16
    layernorm_row<half_t><<<BN_, blk, 0, stream>>>(x, gamma_in, xn, DIM_);

    // Q/K/V projections: [4096 x 1024] @ [1024 x 1024], q scaled by D^-1/2
    const dim3 gProj(DIM_/TN, BN_/TM, 1);
    gemm_f16_wmma<<<gProj, blk, 0, stream>>>(xn, 0,0,1, DIM_, wqT, 0,0,1, DIM_,
        nullptr, qt, 0, DIM_, 0, nullptr, nullptr, 0, 0, DIM_, 0.125f, 0);
    gemm_f16_wmma<<<gProj, blk, 0, stream>>>(xn, 0,0,1, DIM_, wkT, 0,0,1, DIM_,
        nullptr, kt, 0, DIM_, 0, nullptr, nullptr, 0, 0, DIM_, 1.0f, 0);
    gemm_f16_wmma<<<gProj, blk, 0, stream>>>(xn, 0,0,1, DIM_, wvT, 0,0,1, DIM_,
        nullptr, vt, 0, DIM_, 0, nullptr, nullptr, 0, 0, DIM_, 1.0f, 0);

    // per-head transposed V for use as B operand
    split_heads_T<<<nElemPH, blk, 0, stream>>>(vt, vT);

    // S = q @ k^T per (b,h); skip fully-masked blocks  (B = k rows, K-contiguous)
    gemm_f16_wmma<<<dim3(N_/TN, N_/TM, BH_), blk, 0, stream>>>(
        qt, sQKV, D_, H_, DIM_,  kt, sQKV, D_, H_, DIM_,
        nullptr, Sh, sNN, N_, 0, nullptr, nullptr, 0, 0, D_, 1.0f, 1);

    // causal softmax + deg
    softmax_causal<<<BH_*N_, blk, 0, stream>>>(Sh, Ph, deg, N_);

    // attention out = P @ v   -> ao (f32 [bh,N,D]);  B = vT [bh,D,N]
    gemm_f16_wmma<<<dim3(1, N_/TM, BH_), blk, 0, stream>>>(
        Ph, sNN, 0, 1, N_,  vT, (long long)D_*N_, 0, 1, N_,
        ao, nullptr, sND, D_, 0, nullptr, nullptr, 0, 0, N_, 1.0f, 0);

    // GNN: feats0 = v (per head layout)
    split_heads<<<nElemPH, blk, 0, stream>>>(vt, feats);
    for (int l = 0; l < 2; ++l) {
        // hT = (feats @ gnn_w[l] + gnn_b[l]) stored transposed [bh,D,N]
        gemm_f16_wmma<<<dim3(1, N_/TM, BH_), blk, 0, stream>>>(
            feats, sND, 0, 1, D_,  gwT + (long long)l*D_*D_, 0,0,1, D_,
            nullptr, hT, sND, N_, 1, gnn_b + l*D_, nullptr, 0, 0, D_, 1.0f, 0);
        // G = P @ h   (B = hT, K-contiguous)
        gemm_f16_wmma<<<dim3(1, N_/TM, BH_), blk, 0, stream>>>(
            Ph, sNN, 0, 1, N_,  hT, sND, 0, 1, N_,
            Gbuf, nullptr, sND, D_, 0, nullptr, nullptr, 0, 0, N_, 1.0f, 0);
        // feats = gelu((G + (deg-1)*h)/deg + feats)
        gnn_pointwise<<<nElemPH, blk, 0, stream>>>(Gbuf, hT, deg, feats);
    }
    // ao = attention_out + feats @ lin_w + lin_b   (addend = ao, written in place)
    gemm_f16_wmma<<<dim3(1, N_/TM, BH_), blk, 0, stream>>>(
        feats, sND, 0, 1, D_,  lwT, 0,0,1, D_,
        ao, nullptr, sND, D_, 0, lin_b, ao, sND, D_, D_, 1.0f, 0);

    // merge heads -> [BN, 1024] f16, project with Wo, final layernorm
    merge_heads<<<nElemPH, blk, 0, stream>>>(ao, merged);
    gemm_f16_wmma<<<dim3(DIM_/TN, BN_/TM, 1), blk, 0, stream>>>(
        merged, 0,0,1, DIM_,  woT, 0,0,1, DIM_,
        Y, nullptr, 0, DIM_, 0, nullptr, nullptr, 0, 0, DIM_, 1.0f, 0);
    layernorm_row<float><<<BN_, blk, 0, stream>>>(Y, gamma_out, out, DIM_);
}
